// LogisticMapDenseLayer_28613072126791
// MI455X (gfx1250) — matrix operand order repaired and checked by reading, not compiled, not run on hardware
//
#include <hip/hip_runtime.h>

// ---------------------------------------------------------------------------
// LogisticMapDenseLayer for MI455X (gfx1250, wave32, WMMA).
//
// Phase 1 (serial, latency-bound): one lane iterates the logistic map
//   c_{n+1} = (r*c_n) * (1 - c_n)      (exact f32 op order, no FMA fusion)
// and stores each iterate as f16 into a TRANSPOSED weight matrix wT[U][D]
// in the workspace, so the GEMM reads contiguous-K B tiles.
//
// Phase 2 (roofline-bound GEMM): out = x @ w + bias using
// v_wmma_f32_16x16x32_f16 with f32 accumulation. x converted f32->f16 on the
// fly while staging to LDS. 128x64 block tile, 8 waves, 32x32 per wave.
// Double-buffered LDS: one workgroup barrier per K step; next tile's global
// loads + L2 prefetch overlap the current tile's WMMAs.
// ---------------------------------------------------------------------------

typedef _Float16 v16h __attribute__((ext_vector_type(16)));
typedef _Float16 v8h  __attribute__((ext_vector_type(8)));
typedef float    v8f  __attribute__((ext_vector_type(8)));

constexpr int kB = 32768;   // batch
constexpr int kD = 512;     // in features (K)
constexpr int kU = 512;     // units (N)

constexpr int MB = 128;     // block tile M
constexpr int NB = 64;      // block tile N
constexpr int KB = 32;      // K step (one WMMA K)

// ---------------------------------------------------------------------------
// Phase 1: serial logistic-map weight generation (chaotic -> must bit-match
// the reference's mul/sub/mul f32 sequence; __f*_rn blocks contraction).
// ---------------------------------------------------------------------------
__global__ __launch_bounds__(32)
void logistic_gen_wT(const float* __restrict__ r_p,
                     const float* __restrict__ x0_p,
                     _Float16* __restrict__ wT) {
  if (threadIdx.x != 0) return;
  // r_mapped = 3.57 + (4.0 - 3.57) * r, evaluated in f32 like jnp does.
  const float r = 3.57f + 0.43f * r_p[0];
  float c = x0_p[0];
  for (int d = 0; d < kD; ++d) {
    for (int u = 0; u < kU; ++u) {
      const float t0 = __fmul_rn(r, c);       // r*c
      const float t1 = __fsub_rn(1.0f, c);    // 1-c (independent, co-issues)
      c = __fmul_rn(t0, t1);                  // dependent chain: 2 muls/step
      wT[u * kD + d] = (_Float16)c;           // store transposed, f16
    }
  }
}

// ---------------------------------------------------------------------------
// Phase 2: WMMA GEMM. grid = (B/128)*(U/64) blocks, 256 threads (8 waves).
// ---------------------------------------------------------------------------
__global__ __launch_bounds__(256)
void logistic_dense_wmma(const float* __restrict__ x,
                         const _Float16* __restrict__ wT,
                         const float* __restrict__ bias,
                         float* __restrict__ out) {
  __shared__ _Float16 As[2][MB * KB];   // [128][32] f16, row-major, 2x8 KB
  __shared__ _Float16 Bs[2][NB * KB];   // [64][32]  f16, [N][K],   2x4 KB

  const int tid  = threadIdx.x;
  const int lane = tid & 31;
  const int wave = tid >> 5;
  const int wm   = wave & 3;          // wave's 32-row M sub-tile
  const int wn   = wave >> 2;         // wave's 32-col N sub-tile
  const int lr   = lane & 15;
  const bool hi  = lane >= 16;

  const int mb = blockIdx.x >> 3;     // 0..255  (N-fastest -> x reuse in L2)
  const int nb = blockIdx.x & 7;      // 0..7
  const int m0 = mb * MB;
  const int n0 = nb * NB;

  // A staging map: 2 threads per row, 16 f32 each (4x float4).
  const int arow  = tid >> 1;         // 0..127
  const int ahalf = (tid & 1) * 16;   // 0 or 16 (K offset)
  // B staging map: 4 threads per row, 8 f16 (16B) each.
  const int brow = tid >> 2;          // 0..63
  const int bq   = (tid & 3) * 8;     // 0,8,16,24 (K offset)

  const float* xrow = x + (size_t)(m0 + arow) * kD + ahalf;
  const _Float16* wrow = wT + (size_t)(n0 + brow) * kD + bq;

  // Register staging for the in-flight tile.
  float4 a0, a1, a2, a3;
  v8h breg;

  auto load_tile = [&](int k0) {
    const float4* ag = (const float4*)(xrow + k0);
    a0 = ag[0]; a1 = ag[1]; a2 = ag[2]; a3 = ag[3];
    breg = *(const v8h*)(wrow + k0);
  };
  auto store_tile = [&](int buf) {
    const v8h h0 = {(_Float16)a0.x, (_Float16)a0.y, (_Float16)a0.z,
                    (_Float16)a0.w, (_Float16)a1.x, (_Float16)a1.y,
                    (_Float16)a1.z, (_Float16)a1.w};
    const v8h h1 = {(_Float16)a2.x, (_Float16)a2.y, (_Float16)a2.z,
                    (_Float16)a2.w, (_Float16)a3.x, (_Float16)a3.y,
                    (_Float16)a3.z, (_Float16)a3.w};
    *(v8h*)(As[buf] + arow * KB + ahalf)     = h0;   // ds_store_b128
    *(v8h*)(As[buf] + arow * KB + ahalf + 8) = h1;   // ds_store_b128
    *(v8h*)(Bs[buf] + brow * KB + bq)        = breg; // ds_store_b128
  };

  v8f acc[2][2] = {};

  // Prologue: stage tile 0.
  load_tile(0);
  store_tile(0);
  __syncthreads();

  constexpr int NIT = kD / KB;   // 16
  for (int it = 0; it < NIT; ++it) {
    const int cur = it & 1;
    const bool more = (it + 1) < NIT;

    // Kick off next tile's global loads before computing (hides HBM latency),
    // and prefetch two tiles ahead into L2 (global_prefetch_b8).
    if (more) {
      load_tile((it + 1) * KB);
      if (it + 2 < NIT) __builtin_prefetch(xrow + (it + 2) * KB, 0, 1);
    }

    // ---- build fragments per ISA 7.12.2 layouts and issue 4 WMMAs ----
    v16h afrag[2];
#pragma unroll
    for (int sm = 0; sm < 2; ++sm) {
      // A 16x32 f16: lanes<16 hold K {0..7,16..23}; lanes>=16 {8..15,24..31}
      const _Float16* ap =
          As[cur] + (wm * 32 + sm * 16 + lr) * KB + (hi ? 8 : 0);
      const v8h alo = *(const v8h*)ap;          // K base+0..7
      const v8h ahi = *(const v8h*)(ap + 16);   // K base+16..23
      afrag[sm] = __builtin_shufflevector(alo, ahi, 0, 1, 2, 3, 4, 5, 6, 7, 8,
                                          9, 10, 11, 12, 13, 14, 15);
    }
#pragma unroll
    for (int sn = 0; sn < 2; ++sn) {
      // B 32x16 f16: N = lane%16; lanes<16 hold K {0..7,16..23}; lanes>=16 {8..15,24..31}
      const _Float16* bp =
          Bs[cur] + (wn * 32 + sn * 16 + lr) * KB + (hi ? 8 : 0);
      const v8h blo = *(const v8h*)bp;          // K base+0..7
      const v8h bhi = *(const v8h*)(bp + 16);   // K base+16..23
      const v16h bfrag = __builtin_shufflevector(blo, bhi, 0, 1, 2, 3, 4, 5, 6,
                                                 7, 8, 9, 10, 11, 12, 13, 14,
                                                 15);
#pragma unroll
      for (int sm = 0; sm < 2; ++sm) {
        acc[sm][sn] = __builtin_amdgcn_wmma_f32_16x16x32_f16(
            /*neg_a=*/false, afrag[sm], /*neg_b=*/false, bfrag,
            /*c_mod=*/(short)0, acc[sm][sn],
            /*reuse_a=*/false, /*reuse_b=*/false);
      }
    }

    // Stage next tile into the other LDS buffer; single barrier per K step.
    if (more) {
      store_tile(cur ^ 1);
      __syncthreads();
    }
  }

  // ---- epilogue: C/D layout VGPR g -> rows {g, g+8}; col = lane%16 ----
#pragma unroll
  for (int sn = 0; sn < 2; ++sn) {
    const int n = n0 + wn * 32 + sn * 16 + lr;
    const float bv = bias[n];
#pragma unroll
    for (int sm = 0; sm < 2; ++sm) {
      const int mbase = m0 + wm * 32 + sm * 16 + (hi ? 8 : 0);
      float* op = out + (size_t)mbase * kU + n;
#pragma unroll
      for (int g = 0; g < 8; ++g) {
        op[(size_t)g * kU] = acc[sm][sn][g] + bv;
      }
    }
  }
}

// ---------------------------------------------------------------------------
// Inputs (setup_inputs order): x[32768*512] f32, r[1] f32, x0[1] f32,
// bias[512] f32. Output: f32 [32768*512]. Workspace: wT f16 (512 KB).
// ---------------------------------------------------------------------------
extern "C" void kernel_launch(void* const* d_in, const int* in_sizes, int n_in,
                              void* d_out, int out_size, void* d_ws,
                              size_t ws_size, hipStream_t stream) {
  (void)in_sizes; (void)n_in; (void)out_size; (void)ws_size;
  const float* x    = (const float*)d_in[0];
  const float* r    = (const float*)d_in[1];
  const float* x0   = (const float*)d_in[2];
  const float* bias = (const float*)d_in[3];
  float* out        = (float*)d_out;
  _Float16* wT      = (_Float16*)d_ws;   // 512*512*2 B = 512 KB scratch

  hipLaunchKernelGGL(logistic_gen_wT, dim3(1), dim3(32), 0, stream, r, x0, wT);

  const dim3 grid((kB / MB) * (kU / NB));  // 256 * 8 = 2048 blocks
  hipLaunchKernelGGL(logistic_dense_wmma, grid, dim3(256), 0, stream, x, wT,
                     bias, out);
}
